// GNNEncoder_5488968204769
// MI455X (gfx1250) — matrix operand order repaired and checked by reading, not compiled, not run on hardware
//
#include <hip/hip_runtime.h>
#include <hip/hip_bf16.h>

typedef __attribute__((ext_vector_type(16))) __bf16 v16bf;
typedef __attribute__((ext_vector_type(8)))  float  v8f;

#define N_NODES   50000
#define E_RAW     800000
#define E_TOT     850000
#define NEG_SLOPE 0.2f
#define COUT      128

// ---------------- helpers ----------------

__device__ __forceinline__ void atomicMaxF32(float* addr, float v) {
    // ordered-int trick: works for mixed signs with -inf init
    if (v >= 0.0f) atomicMax((int*)addr, __float_as_int(v));
    else           atomicMin((unsigned int*)addr, __float_as_uint(v));
}

__device__ __forceinline__ void edge_sd(int e, const int* __restrict__ ei, int& s, int& d) {
    if (e < E_RAW) { s = ei[e]; d = ei[E_RAW + e]; }
    else           { s = e - E_RAW; d = s; }
}

// ---------------- tiny fold: v[k] = sum_j w[k,j] * att[j] ----------------

__global__ void fold_kernel(const float* __restrict__ w, const float* __restrict__ att,
                            float* __restrict__ v, int K) {
    int k = blockIdx.x * blockDim.x + threadIdx.x;
    if (k >= K) return;
    float s = 0.f;
    for (int j = 0; j < COUT; ++j) s += w[k * COUT + j] * att[j];
    v[k] = s;
}

// ---------------- rowwise dot: out[n] = A[n,:] . v  (1 wave / row) ----------------

__global__ void rowdot_kernel(const float* __restrict__ A, const float* __restrict__ v,
                              float* __restrict__ out, int K) {
    int wid  = (blockIdx.x * blockDim.x + threadIdx.x) >> 5;
    int lane = threadIdx.x & 31;
    if (wid >= N_NODES) return;
    const float* row = A + (size_t)wid * K;
    float s = 0.f;
    for (int k = lane * 4; k < K; k += 128) {
        float4 f = *(const float4*)(row + k);
        s += f.x * v[k] + f.y * v[k + 1] + f.z * v[k + 2] + f.w * v[k + 3];
    }
    for (int off = 16; off > 0; off >>= 1) s += __shfl_xor(s, off, 32);
    if (lane == 0) out[wid] = s;
}

// ---------------- WMMA GEMM: C[N_NODES,128] = A[N_NODES,K] @ W[K,128] (bf16 math, f32 acc)
// 128 threads = 4 waves; each wave owns a 16x128 tile (8 x v_wmma_f32_16x16x32_bf16 per K-step)

__global__ void __launch_bounds__(128)
gemm_wmma_kernel(const float* __restrict__ A, const float* __restrict__ W,
                 float* __restrict__ C, int K) {
    // B fragments pre-swizzled to the ISA 16-bit B 32x16 layout:
    // slot = (kt*8 + nt)*32 + lane ; lane&15 -> N column, lane half h -> K {8h..8h+7, 16+8h..16+8h+7}
    __shared__ __bf16 wlds[256 * COUT];   // 64 KB max (K<=256)

    const int tid  = threadIdx.x;
    const int lane = tid & 31;
    const int wave = tid >> 5;
    const int nkt  = K >> 5;

    for (int slot = tid; slot < nkt * 8 * 32; slot += 128) {
        int ls = slot & 31;
        int nt = (slot >> 5) & 7;
        int kt = slot >> 8;
        int n  = nt * 16 + (ls & 15);
        int kb = kt * 32 + ((ls < 16) ? 0 : 8);
        __bf16* dstp = &wlds[slot * 16];
        #pragma unroll
        for (int i = 0; i < 16; ++i)
            dstp[i] = (__bf16)W[(kb + ((i < 8) ? i : i + 8)) * COUT + n];
    }
    __syncthreads();

    const int rowBase = blockIdx.x * 64 + wave * 16;
    int m   = lane & 15;
    int row = rowBase + m;
    if (row > N_NODES - 1) row = N_NODES - 1;          // clamp loads, guard stores
    const float* arow  = A + (size_t)row * K;
    const int    kbase = (lane < 16) ? 0 : 8;          // A layout K offsets per lane half

    v8f acc[8];
    #pragma unroll
    for (int nt = 0; nt < 8; ++nt) acc[nt] = 0.0f;

    for (int kt = 0; kt < nkt; ++kt) {
        const int k0 = kt * 32;
        float4 f0 = *(const float4*)(arow + k0 + kbase);
        float4 f1 = *(const float4*)(arow + k0 + kbase + 4);
        float4 f2 = *(const float4*)(arow + k0 + kbase + 16);
        float4 f3 = *(const float4*)(arow + k0 + kbase + 20);
        v16bf a;
        a[0]  = (__bf16)f0.x; a[1]  = (__bf16)f0.y; a[2]  = (__bf16)f0.z; a[3]  = (__bf16)f0.w;
        a[4]  = (__bf16)f1.x; a[5]  = (__bf16)f1.y; a[6]  = (__bf16)f1.z; a[7]  = (__bf16)f1.w;
        a[8]  = (__bf16)f2.x; a[9]  = (__bf16)f2.y; a[10] = (__bf16)f2.z; a[11] = (__bf16)f2.w;
        a[12] = (__bf16)f3.x; a[13] = (__bf16)f3.y; a[14] = (__bf16)f3.z; a[15] = (__bf16)f3.w;
        #pragma unroll
        for (int nt = 0; nt < 8; ++nt) {
            v16bf b = *(const v16bf*)&wlds[((kt * 8 + nt) * 32 + lane) * 16];
            acc[nt] = __builtin_amdgcn_wmma_f32_16x16x32_bf16(
                false, a, false, b, (short)0, acc[nt], false, false);
        }
    }

    // C/D layout: lane&15 -> N, lane half selects M rows {r} vs {8+r}
    const int nlo = lane & 15;
    const int mhi = (lane >> 4) * 8;
    #pragma unroll
    for (int r = 0; r < 8; ++r) {
        int orow = rowBase + mhi + r;
        if (orow < N_NODES) {
            float* op = C + (size_t)orow * COUT + nlo;
            #pragma unroll
            for (int nt = 0; nt < 8; ++nt) op[nt * 16] = acc[nt][r];
        }
    }
}

// ---------------- per-node init ----------------

__global__ void init_seg_kernel(float* __restrict__ mseg, float* __restrict__ denom) {
    int n = blockIdx.x * blockDim.x + threadIdx.x;
    if (n >= N_NODES) return;
    mseg[n]  = -__builtin_inff();
    denom[n] = 0.f;
}

__global__ void init_bias_kernel(float* __restrict__ out, const float* __restrict__ bias) {
    int i = blockIdx.x * blockDim.x + threadIdx.x;
    if (i >= N_NODES * COUT) return;
    out[i] = bias[i & (COUT - 1)];
}

__global__ void relu_kernel(float* __restrict__ h) {
    int i = blockIdx.x * blockDim.x + threadIdx.x;
    if (i >= N_NODES * COUT) return;
    float v = h[i];
    h[i] = v > 0.f ? v : 0.f;
}

// ---------------- edge passes ----------------

__global__ void edge_logit_kernel(const int* __restrict__ ei,
                                  const float* __restrict__ a_s, const float* __restrict__ a_d,
                                  float* __restrict__ elog, float* __restrict__ mseg) {
    int e = blockIdx.x * blockDim.x + threadIdx.x;
    if (e >= E_TOT) return;
    int s, d; edge_sd(e, ei, s, d);
    float v = a_s[s] + a_d[d];
    v = (v > 0.f) ? v : NEG_SLOPE * v;
    elog[e] = v;
    atomicMaxF32(&mseg[d], v);
}

__global__ void edge_exp_kernel(const int* __restrict__ ei,
                                const float* __restrict__ elog, const float* __restrict__ mseg,
                                float* __restrict__ eexp, float* __restrict__ denom) {
    int e = blockIdx.x * blockDim.x + threadIdx.x;
    if (e >= E_TOT) return;
    int d = (e < E_RAW) ? ei[E_RAW + e] : (e - E_RAW);
    float ex = __expf(elog[e] - mseg[d]);
    eexp[e] = ex;
    atomicAdd(&denom[d], ex);
}

// 1 wave per edge: gather xs[src] (float4/lane), scatter-add alpha*xs into out[dst]
__global__ void edge_aggr_kernel(const int* __restrict__ ei,
                                 const float* __restrict__ eexp, const float* __restrict__ denom,
                                 const float* __restrict__ xs, float* __restrict__ out) {
    int wid  = (blockIdx.x * blockDim.x + threadIdx.x) >> 5;
    int lane = threadIdx.x & 31;
    if (wid >= E_TOT) return;
    int s, d; edge_sd(wid, ei, s, d);
    float alpha = eexp[wid] / denom[d];
    float4 v = *(const float4*)(xs + (size_t)s * COUT + lane * 4);
    float* o = out + (size_t)d * COUT + lane * 4;
    atomicAdd(o + 0, alpha * v.x);
    atomicAdd(o + 1, alpha * v.y);
    atomicAdd(o + 2, alpha * v.z);
    atomicAdd(o + 3, alpha * v.w);
}

// ---------------- launch ----------------

extern "C" void kernel_launch(void* const* d_in, const int* in_sizes, int n_in,
                              void* d_out, int out_size, void* d_ws, size_t ws_size,
                              hipStream_t stream) {
    const float* x      = (const float*)d_in[0];
    const int*   ei     = (const int*)  d_in[1];
    const float* w1_src = (const float*)d_in[2];
    const float* w1_dst = (const float*)d_in[3];
    const float* att1_s = (const float*)d_in[4];
    const float* att1_d = (const float*)d_in[5];
    const float* b1     = (const float*)d_in[6];
    const float* w2_src = (const float*)d_in[7];
    const float* w2_dst = (const float*)d_in[8];
    const float* att2_s = (const float*)d_in[9];
    const float* att2_d = (const float*)d_in[10];
    const float* b2     = (const float*)d_in[11];
    float* out = (float*)d_out;

    float* ws    = (float*)d_ws;
    float* xs    = ws; ws += (size_t)N_NODES * COUT;   // current layer messages
    float* h     = ws; ws += (size_t)N_NODES * COUT;   // layer-1 output / layer-2 input
    float* a_s   = ws; ws += N_NODES;
    float* a_d   = ws; ws += N_NODES;
    float* mseg  = ws; ws += N_NODES;
    float* denom = ws; ws += N_NODES;
    float* elog  = ws; ws += E_TOT;
    float* eexp  = ws; ws += E_TOT;
    float* vfold = ws; ws += 256;

    const dim3 b256(256);
    const int gNodes = (N_NODES + 255) / 256;
    const int gEdges = (E_TOT + 255) / 256;
    const int gWaveN = (N_NODES * 32 + 255) / 256;
    const int gWaveE = ((size_t)E_TOT * 32 + 255) / 256;
    const int gElems = (N_NODES * COUT + 255) / 256;
    const int gGemm  = (N_NODES + 63) / 64;

    // -------- layer 1 (d_in=256) --------
    fold_kernel<<<1, b256, 0, stream>>>(w1_dst, att1_d, vfold, 256);
    gemm_wmma_kernel<<<gGemm, 128, 0, stream>>>(x, w1_src, xs, 256);
    rowdot_kernel<<<gWaveN, b256, 0, stream>>>(xs, att1_s, a_s, COUT);
    rowdot_kernel<<<gWaveN, b256, 0, stream>>>(x, vfold, a_d, 256);
    init_seg_kernel<<<gNodes, b256, 0, stream>>>(mseg, denom);
    init_bias_kernel<<<gElems, b256, 0, stream>>>(h, b1);
    edge_logit_kernel<<<gEdges, b256, 0, stream>>>(ei, a_s, a_d, elog, mseg);
    edge_exp_kernel<<<gEdges, b256, 0, stream>>>(ei, elog, mseg, eexp, denom);
    edge_aggr_kernel<<<gWaveE, b256, 0, stream>>>(ei, eexp, denom, xs, h);
    relu_kernel<<<gElems, b256, 0, stream>>>(h);

    // -------- layer 2 (hid=128) --------
    fold_kernel<<<1, b256, 0, stream>>>(w2_dst, att2_d, vfold, 128);
    gemm_wmma_kernel<<<gGemm, 128, 0, stream>>>(h, w2_src, xs, 128);
    rowdot_kernel<<<gWaveN, b256, 0, stream>>>(xs, att2_s, a_s, COUT);
    rowdot_kernel<<<gWaveN, b256, 0, stream>>>(h, vfold, a_d, COUT);
    init_seg_kernel<<<gNodes, b256, 0, stream>>>(mseg, denom);
    init_bias_kernel<<<gElems, b256, 0, stream>>>(out, b2);
    edge_logit_kernel<<<gEdges, b256, 0, stream>>>(ei, a_s, a_d, elog, mseg);
    edge_exp_kernel<<<gEdges, b256, 0, stream>>>(ei, elog, mseg, eexp, denom);
    edge_aggr_kernel<<<gWaveE, b256, 0, stream>>>(ei, eexp, denom, xs, out);
}
